// TED_25606595019057
// MI455X (gfx1250) — matrix operand order repaired and checked by reading, not compiled, not run on hardware
//
#include <hip/hip_runtime.h>
#include <cstdint>
#include <cstddef>

#define DIMX 512
#define HIDX 2048
#define NLAYER 4
#define VOCABX 32000
#define BBX 2
#define SSX 2048
#define MTOK (BBX*SSX)   // 4096 tokens

typedef __bf16 bf16;
typedef __attribute__((ext_vector_type(16))) __bf16 v16bf;
typedef __attribute__((ext_vector_type(8)))  __bf16 v8bf;
typedef __attribute__((ext_vector_type(4)))  __bf16 v4bf;
typedef __attribute__((ext_vector_type(8)))  float  v8f;
typedef __attribute__((ext_vector_type(4)))  int    v4i;

// ---------- CDNA5 async global->LDS copy (16B per lane, ASYNCcnt) ------------
__device__ __forceinline__ void async_b128(const bf16* g, bf16* l) {
#if __has_builtin(__builtin_amdgcn_global_load_async_to_lds_b128)
  __builtin_amdgcn_global_load_async_to_lds_b128(
      (__attribute__((address_space(1))) v4i*)g,
      (__attribute__((address_space(3))) v4i*)l, 0, 0);
#else
  unsigned loff = (unsigned)(unsigned long long)
      (const __attribute__((address_space(3))) bf16*)l;
  asm volatile("global_load_async_to_lds_b128 %0, %1, off"
               :: "v"(loff), "v"(g) : "memory");
#endif
}
__device__ __forceinline__ void wait_async_zero() {
#if __has_builtin(__builtin_amdgcn_s_wait_asynccnt)
  __builtin_amdgcn_s_wait_asynccnt(0);
#else
  asm volatile("s_wait_asynccnt 0x0" ::: "memory");
#endif
}

// ---------------- embedding gather: x[row,:] = emb[tok[row],:] ----------------
__global__ __launch_bounds__(128)
void gather_kernel(const int* __restrict__ tok, const float* __restrict__ emb,
                   float* __restrict__ x) {
  const int row = blockIdx.x;
  const int t = tok[row];
  const float4* src = (const float4*)(emb + (size_t)t * DIMX);
  float4* dst = (float4*)(x + (size_t)row * DIMX);
  dst[threadIdx.x] = src[threadIdx.x];          // DIMX/4 == 128 threads
}

// ---------------- block reduction helper (256 threads, wave32) ----------------
__device__ __forceinline__ float block_sum(float v, float* sbuf) {
  #pragma unroll
  for (int off = 16; off > 0; off >>= 1) v += __shfl_down(v, off, 32);
  const int wid = threadIdx.x >> 5;
  if ((threadIdx.x & 31) == 0) sbuf[wid] = v;
  __syncthreads();
  float r = (threadIdx.x < 8) ? sbuf[threadIdx.x] : 0.f;
  #pragma unroll
  for (int off = 4; off > 0; off >>= 1) r += __shfl_down(r, off, 32);
  if (threadIdx.x == 0) sbuf[0] = r;
  __syncthreads();
  const float out = sbuf[0];
  __syncthreads();
  return out;
}

// ------- fused rmsnorm (+ optional lambda dot -> log_sigmoid) -> bf16 ---------
__global__ __launch_bounds__(256)
void rmsnorm_kernel(const float* __restrict__ x, const float* __restrict__ w,
                    const float* __restrict__ lam, bf16* __restrict__ out_bf,
                    float* __restrict__ nl_out) {
  __shared__ float sbuf[8];
  const int row = blockIdx.x;
  const float* xr = x + (size_t)row * DIMX;
  const float x0 = xr[threadIdx.x], x1 = xr[threadIdx.x + 256];
  const float ss = block_sum(x0 * x0 + x1 * x1, sbuf);
  const float rstd = rsqrtf(ss * (1.0f / DIMX) + 1e-6f);
  const float h0 = x0 * rstd * w[threadIdx.x];
  const float h1 = x1 * rstd * w[threadIdx.x + 256];
  out_bf[(size_t)row * DIMX + threadIdx.x]       = (bf16)h0;
  out_bf[(size_t)row * DIMX + threadIdx.x + 256] = (bf16)h1;
  if (lam) {
    const float d = h0 * lam[threadIdx.x] + h1 * lam[threadIdx.x + 256];
    const float z = block_sum(d, sbuf);
    if (threadIdx.x == 0)
      nl_out[row] = fminf(z, 0.f) - log1pf(__expf(-fabsf(z)));  // log_sigmoid
  }
}

// --------------- elementwise: m = h * silu(g), f32 -> bf16 --------------------
__global__ __launch_bounds__(256)
void silu_mul_kernel(const float* __restrict__ hc, const float* __restrict__ gc,
                     bf16* __restrict__ mbf) {
  const size_t i = ((size_t)blockIdx.x * 256 + threadIdx.x) * 4;
  const float4 a = *(const float4*)(hc + i);
  const float4 g = *(const float4*)(gc + i);
  v4bf o;
  o[0] = (bf16)(a.x * (g.x / (1.f + __expf(-g.x))));
  o[1] = (bf16)(a.y * (g.y / (1.f + __expf(-g.y))));
  o[2] = (bf16)(a.z * (g.z / (1.f + __expf(-g.z))));
  o[3] = (bf16)(a.w * (g.w / (1.f + __expf(-g.w))));
  *(v4bf*)(mbf + i) = o;
}

// ---------------- WMMA fragment builders (ISA 7.12.2, wave32) -----------------
__device__ __forceinline__ v16bf frag16(const bf16* lo_p, const bf16* hi_p) {
  const v8bf lo = *(const v8bf*)lo_p;
  const v8bf hi = *(const v8bf*)hi_p;
  v16bf r;
  #pragma unroll
  for (int e = 0; e < 8; ++e) { r[e] = lo[e]; r[e + 8] = hi[e]; }
  return r;
}
__device__ __forceinline__ void packW16(float4 a, float4 b, float4 c, float4 d,
                                        bf16* lW) {
  v8bf p0, p1;
  p0[0] = (bf16)a.x; p0[1] = (bf16)a.y; p0[2] = (bf16)a.z; p0[3] = (bf16)a.w;
  p0[4] = (bf16)b.x; p0[5] = (bf16)b.y; p0[6] = (bf16)b.z; p0[7] = (bf16)b.w;
  p1[0] = (bf16)c.x; p1[1] = (bf16)c.y; p1[2] = (bf16)c.z; p1[3] = (bf16)c.w;
  p1[4] = (bf16)d.x; p1[5] = (bf16)d.y; p1[6] = (bf16)d.z; p1[7] = (bf16)d.w;
  *(v8bf*)lW = p0; *(v8bf*)(lW + 8) = p1;
}

// ------------------------- WMMA bf16 GEMM: out = A @ W^T ----------------------
// A: [M,K] bf16 (activations, async-copied to LDS). W: [N,K] f32 (weights),
// converted f32->bf16 while staging. Double-buffered; 128x64 block tile, BK=64.
// MODE 0: out=acc; 1: out+=acc; 2: out += acc * aux; 3: out=acc nontemporal.
template <int MODE>
__global__ __launch_bounds__(256)
void gemm_kernel(const bf16* __restrict__ A, const float* __restrict__ W,
                 float* __restrict__ out, const float* __restrict__ aux,
                 int N, int K) {
  constexpr int ASR = 72, WSR = 72;       // LDS row strides (144B, 16B aligned)
  __shared__ bf16 ldsA[2][128 * ASR];
  __shared__ bf16 ldsW[2][64 * WSR];
  const int rowBase = blockIdx.y * 128;
  const int colBase = blockIdx.x * 64;
  const int tid = threadIdx.x;
  const int wave = tid >> 5, wm = wave & 3, wn = wave >> 2;  // 4x2 waves
  const int lane = tid & 31, half = lane >> 4, lm = lane & 15;

  v8f acc[2][2];
  #pragma unroll
  for (int i = 0; i < 2; ++i)
    #pragma unroll
    for (int j = 0; j < 2; ++j)
      acc[i][j] = (v8f){0.f, 0.f, 0.f, 0.f, 0.f, 0.f, 0.f, 0.f};

  const int ar = tid >> 1, ac = (tid & 1) * 32;   // A loader: 64B per thread
  const int wr = tid >> 2, wc = (tid & 3) * 16;   // W loader: 16 f32 per thread
  const int nk = K >> 6;                          // K multiple of 64

  // ---- prologue: stage tile 0 ----
  {
    const bf16* gA = A + (size_t)(rowBase + ar) * K + ac;
    bf16* lA = &ldsA[0][ar * ASR + ac];
    async_b128(gA, lA);           async_b128(gA + 8,  lA + 8);
    async_b128(gA + 16, lA + 16); async_b128(gA + 24, lA + 24);
    const float* gW = W + (size_t)(colBase + wr) * K + wc;
    const float4 w0 = ((const float4*)gW)[0], w1 = ((const float4*)gW)[1];
    const float4 w2 = ((const float4*)gW)[2], w3 = ((const float4*)gW)[3];
    packW16(w0, w1, w2, w3, &ldsW[0][wr * WSR + wc]);
    wait_async_zero();
  }
  __syncthreads();

  for (int kt = 0; kt < nk; ++kt) {
    const int cur = kt & 1, nxt = cur ^ 1;
    const bool more = (kt + 1 < nk);
    float4 w0, w1, w2, w3;
    if (more) {   // issue next tile: async A -> LDS, W -> regs (no wait yet)
      const bf16* gA = A + (size_t)(rowBase + ar) * K + (kt + 1) * 64 + ac;
      bf16* lA = &ldsA[nxt][ar * ASR + ac];
      async_b128(gA, lA);           async_b128(gA + 8,  lA + 8);
      async_b128(gA + 16, lA + 16); async_b128(gA + 24, lA + 24);
      const float* gW = W + (size_t)(colBase + wr) * K + (kt + 1) * 64 + wc;
      w0 = ((const float4*)gW)[0]; w1 = ((const float4*)gW)[1];
      w2 = ((const float4*)gW)[2]; w3 = ((const float4*)gW)[3];
      if (kt + 2 < nk)
        __builtin_prefetch(W + (size_t)(colBase + wr) * K + (kt + 2) * 64 + wc, 0, 1);
    }
    // ---- compute on current buffers: 2 k-steps x 4 WMMAs ----
    #pragma unroll
    for (int ks = 0; ks < 2; ++ks) {
      v16bf af[2], bfg[2];
      #pragma unroll
      for (int i = 0; i < 2; ++i) {
        const bf16* ap = &ldsA[cur][(wm * 32 + i * 16 + lm) * ASR + ks * 32];
        af[i] = frag16(ap + half * 16, ap + half * 16 + 8);
      }
      #pragma unroll
      for (int j = 0; j < 2; ++j) {
        const bf16* bp =
            &ldsW[cur][(wn * 32 + j * 16 + lm) * WSR + ks * 32 + half * 16];
        bfg[j] = frag16(bp, bp + 8);
      }
      #pragma unroll
      for (int i = 0; i < 2; ++i)
        #pragma unroll
        for (int j = 0; j < 2; ++j)
          acc[i][j] = __builtin_amdgcn_wmma_f32_16x16x32_bf16(
              false, af[i], false, bfg[j], (short)0, acc[i][j], false, false);
    }
    if (more) {
      packW16(w0, w1, w2, w3, &ldsW[nxt][wr * WSR + wc]);
      wait_async_zero();
    }
    __syncthreads();
  }

  // Epilogue: C/D layout — VGPR r: lanes0-15 M=r, lanes16-31 M=r+8; N=lane%16.
  #pragma unroll
  for (int i = 0; i < 2; ++i) {
    const int rb = rowBase + wm * 32 + i * 16 + 8 * half;
    #pragma unroll
    for (int j = 0; j < 2; ++j) {
      const int col = colBase + wn * 32 + j * 16 + lm;
      #pragma unroll
      for (int r = 0; r < 8; ++r) {
        const size_t o = (size_t)(rb + r) * N + col;
        if (MODE == 0)      out[o] = acc[i][j][r];
        else if (MODE == 1) out[o] += acc[i][j][r];
        else if (MODE == 2) out[o] += acc[i][j][r] * aux[o];
        else __builtin_nontemporal_store(acc[i][j][r], &out[o]);  // streamed
      }
    }
  }
}

// ---- causal decay mix as WMMA GEMM: out = D @ q, D built on the fly ----------
// D[s,t] = exp(nl[t]*(s-t)) for t<=s else 0.  Double-buffered; branchless mask.
// Epilogue: silu -> bf16.
__global__ __launch_bounds__(256)
void decay_wmma_kernel(const float* __restrict__ q, const float* __restrict__ nl,
                       bf16* __restrict__ sobf) {
  constexpr int DSR = 40, QSR = 40;
  __shared__ bf16 ldsD[2][128 * DSR];   // [s 128][t 32]
  __shared__ bf16 ldsQ[2][64 * QSR];    // [d 64][t 32]  (transposed q tile)
  const int b = blockIdx.z;
  const int sBase = blockIdx.y * 128;
  const int dBase = blockIdx.x * 64;
  const float* nlb = nl + (size_t)b * SSX;
  const float* qb  = q  + (size_t)b * SSX * DIMX;
  const int tid = threadIdx.x;
  const int wave = tid >> 5, wm = wave & 3, wn = wave >> 2;
  const int lane = tid & 31, half = lane >> 4, lm = lane & 15;

  v8f acc[2][2];
  #pragma unroll
  for (int i = 0; i < 2; ++i)
    #pragma unroll
    for (int j = 0; j < 2; ++j)
      acc[i][j] = (v8f){0.f, 0.f, 0.f, 0.f, 0.f, 0.f, 0.f, 0.f};

  const int nk = (sBase >> 5) + 4;              // t-chunks cover t <= sBase+127
  const int sr = tid >> 1, tc0 = (tid & 1) * 16;     // D builder coords
  const int s_g = sBase + sr;
  const int tq = tid >> 3, dq = (tid & 7) * 8;       // Q loader coords

  // unconditional vector loads of nl (always in-bounds: t <= sBase+127 < S)
  auto loadNl = [&](int tBase, float* nv) {
    const float* np = nlb + tBase + tc0;
    const float4 a = ((const float4*)np)[0], c = ((const float4*)np)[1];
    const float4 e = ((const float4*)np)[2], g = ((const float4*)np)[3];
    nv[0] = a.x; nv[1] = a.y; nv[2]  = a.z; nv[3]  = a.w;
    nv[4] = c.x; nv[5] = c.y; nv[6]  = c.z; nv[7]  = c.w;
    nv[8] = e.x; nv[9] = e.y; nv[10] = e.z; nv[11] = e.w;
    nv[12] = g.x; nv[13] = g.y; nv[14] = g.z; nv[15] = g.w;
  };
  auto loadQ = [&](int tBase, float4& q0, float4& q1) {
    const float* qr = qb + (size_t)(tBase + tq) * DIMX + dBase + dq;
    q0 = ((const float4*)qr)[0]; q1 = ((const float4*)qr)[1];
  };
  // branchless: exp computed unconditionally, causal mask via select
  auto storeD = [&](int tBase, int buf, const float* nv) {
    v8bf p0, p1;
    #pragma unroll
    for (int j = 0; j < 8; ++j) {
      const int t0 = tBase + tc0 + j, t1 = t0 + 8;
      const float d0 = (float)(s_g - t0), d1 = (float)(s_g - t1);
      float e0 = __expf(nv[j] * d0);
      float e1 = __expf(nv[j + 8] * d1);
      e0 = (d0 >= 0.f) ? e0 : 0.f;     // v_cndmask, no EXEC branches
      e1 = (d1 >= 0.f) ? e1 : 0.f;
      p0[j] = (bf16)e0; p1[j] = (bf16)e1;
    }
    bf16* dp = &ldsD[buf][sr * DSR + tc0];
    *(v8bf*)dp = p0; *(v8bf*)(dp + 8) = p1;
  };
  auto storeQ = [&](int buf, float4 q0, float4 q1) {
    bf16* lq = &ldsQ[buf][0];
    lq[(dq + 0) * QSR + tq] = (bf16)q0.x;
    lq[(dq + 1) * QSR + tq] = (bf16)q0.y;
    lq[(dq + 2) * QSR + tq] = (bf16)q0.z;
    lq[(dq + 3) * QSR + tq] = (bf16)q0.w;
    lq[(dq + 4) * QSR + tq] = (bf16)q1.x;
    lq[(dq + 5) * QSR + tq] = (bf16)q1.y;
    lq[(dq + 6) * QSR + tq] = (bf16)q1.z;
    lq[(dq + 7) * QSR + tq] = (bf16)q1.w;
  };

  // ---- prologue: build chunk 0 ----
  {
    float nv[16]; float4 q0, q1;
    loadNl(0, nv); loadQ(0, q0, q1);
    storeD(0, 0, nv); storeQ(0, q0, q1);
  }
  __syncthreads();

  for (int kt = 0; kt < nk; ++kt) {
    const int cur = kt & 1, nxt = cur ^ 1;
    const bool more = (kt + 1 < nk);
    float nv[16]; float4 q0, q1;
    if (more) { loadNl((kt + 1) * 32, nv); loadQ((kt + 1) * 32, q0, q1); }
    // ---- 4 WMMAs on current buffers ----
    v16bf af[2], bfg[2];
    #pragma unroll
    for (int i = 0; i < 2; ++i) {
      const bf16* ap = &ldsD[cur][(wm * 32 + i * 16 + lm) * DSR];
      af[i] = frag16(ap + half * 16, ap + half * 16 + 8);
    }
    #pragma unroll
    for (int j = 0; j < 2; ++j) {
      const bf16* bp = &ldsQ[cur][(wn * 32 + j * 16 + lm) * QSR + half * 16];
      bfg[j] = frag16(bp, bp + 8);
    }
    #pragma unroll
    for (int i = 0; i < 2; ++i)
      #pragma unroll
      for (int j = 0; j < 2; ++j)
        acc[i][j] = __builtin_amdgcn_wmma_f32_16x16x32_bf16(
            false, af[i], false, bfg[j], (short)0, acc[i][j], false, false);
    if (more) { storeD((kt + 1) * 32, nxt, nv); storeQ(nxt, q0, q1); }
    __syncthreads();
  }

  // epilogue: silu, cast bf16 (A-operand of the next GEMM)
  #pragma unroll
  for (int i = 0; i < 2; ++i) {
    const int rs = sBase + wm * 32 + i * 16 + 8 * half;
    #pragma unroll
    for (int j = 0; j < 2; ++j) {
      const int col = dBase + wn * 32 + j * 16 + lm;
      #pragma unroll
      for (int r = 0; r < 8; ++r) {
        const float v = acc[i][j][r];
        sobf[((size_t)b * SSX + rs + r) * DIMX + col] =
            (bf16)(v / (1.f + __expf(-v)));
      }
    }
  }
}

extern "C" void kernel_launch(void* const* d_in, const int* in_sizes, int n_in,
                              void* d_out, int out_size, void* d_ws, size_t ws_size,
                              hipStream_t stream) {
  (void)in_sizes; (void)n_in; (void)out_size; (void)ws_size;
  const int*   tokens       = (const int*)d_in[0];
  const float* emb          = (const float*)d_in[1];
  const float* decay_norm_w = (const float*)d_in[2];
  const float* lambda_w     = (const float*)d_in[3];
  const float* quantity_w   = (const float*)d_in[4];
  const float* gate_w       = (const float*)d_in[5];
  const float* output_w     = (const float*)d_in[6];
  const float* ffn_norm_w   = (const float*)d_in[7];
  const float* w_h          = (const float*)d_in[8];
  const float* w_g          = (const float*)d_in[9];
  const float* w_o          = (const float*)d_in[10];
  const float* out_norm_w   = (const float*)d_in[11];
  float* logits = (float*)d_out;

  // Workspace partitions (~96 MB total, 256B-aligned offsets)
  uint8_t* p  = (uint8_t*)d_ws;
  float* xf   = (float*)(p + 0);            // 4096x512  f32  (residual x)
  bf16*  hbf  = (bf16*) (p + 8388608);      // 4096x512  bf16 (h / f / xnorm)
  float* nl   = (float*)(p + 12582912);     // 4096      f32
  float* qb   = (float*)(p + 12599296);     // 4096x2048 f32  (q / hcand)
  float* gb   = (float*)(p + 46153728);     // 4096x2048 f32  (gate / gcand)
  bf16*  sob  = (bf16*) (p + 79708160);     // 4096x512  bf16 (silu(out))
  bf16*  mbf  = (bf16*) (p + 83902464);     // 4096x2048 bf16 (ffn mid)

  gather_kernel<<<MTOK, 128, 0, stream>>>(tokens, emb, xf);
  for (int l = 0; l < NLAYER; ++l) {
    rmsnorm_kernel<<<MTOK, 256, 0, stream>>>(xf, decay_norm_w + l * DIMX,
                                             lambda_w + l * DIMX, hbf, nl);
    gemm_kernel<0><<<dim3(DIMX / 64, MTOK / 128), 256, 0, stream>>>(
        hbf, quantity_w + (size_t)l * DIMX * DIMX, qb, nullptr, DIMX, DIMX);
    gemm_kernel<0><<<dim3(DIMX / 64, MTOK / 128), 256, 0, stream>>>(
        hbf, gate_w + (size_t)l * DIMX * DIMX, gb, nullptr, DIMX, DIMX);
    decay_wmma_kernel<<<dim3(DIMX / 64, SSX / 128, BBX), 256, 0, stream>>>(
        qb, nl, sob);
    gemm_kernel<2><<<dim3(DIMX / 64, MTOK / 128), 256, 0, stream>>>(
        sob, output_w + (size_t)l * DIMX * DIMX, xf, gb, DIMX, DIMX);
    rmsnorm_kernel<<<MTOK, 256, 0, stream>>>(xf, ffn_norm_w + l * DIMX,
                                             nullptr, hbf, nullptr);
    gemm_kernel<0><<<dim3(HIDX / 64, MTOK / 128), 256, 0, stream>>>(
        hbf, w_h + (size_t)l * HIDX * DIMX, qb, nullptr, HIDX, DIMX);
    gemm_kernel<0><<<dim3(HIDX / 64, MTOK / 128), 256, 0, stream>>>(
        hbf, w_g + (size_t)l * HIDX * DIMX, gb, nullptr, HIDX, DIMX);
    silu_mul_kernel<<<(MTOK * HIDX) / 1024, 256, 0, stream>>>(qb, gb, mbf);
    gemm_kernel<1><<<dim3(DIMX / 64, MTOK / 128), 256, 0, stream>>>(
        mbf, w_o + (size_t)l * DIMX * HIDX, xf, nullptr, DIMX, HIDX);
  }
  rmsnorm_kernel<<<MTOK, 256, 0, stream>>>(xf, out_norm_w, nullptr, hbf, nullptr);
  gemm_kernel<3><<<dim3(VOCABX / 64, MTOK / 128), 256, 0, stream>>>(
      hbf, emb, logits, nullptr, VOCABX, DIMX);
}
